// CityGRU_60447369724128
// MI455X (gfx1250) — compile-verified, matrix operand order checked by reading, not run on hardware
//
#include <hip/hip_runtime.h>
#include <hip/hip_bf16.h>

typedef __attribute__((ext_vector_type(16))) _Float16 v16h;
typedef __attribute__((ext_vector_type(8)))  float    v8f;

#define BATCH 1024
#define HID   1024
#define G3DIM 3072
#define OUTN  50000
#define CITY  50000

#define MT 128
#define NT 128
#define KC 32
#define LDSROW 40   // padded row stride in halves (80B) to spread LDS banks

union H4 { _Float16 h[4]; uint2 u; };

// Load one 16x32 f16 fragment (A or B operand) from an LDS tile laid out
// row-major [rows][LDSROW]. Matches CDNA5 16-bit A-matrix layout:
//   lanes 0-15 : row = lane,     K = 0..7  and 16..23
//   lanes 16-31: row = lane-16,  K = 8..15 and 24..31
__device__ __forceinline__ v16h frag_lds(const _Float16* tile, int rowBase, int lane) {
  const int sub = lane >> 4;                 // 0 or 1
  const int r   = rowBase + (lane & 15);
  const _Float16* p = tile + r * LDSROW + sub * 8;
  union { float4 f[2]; v16h v; } u;
  u.f[0] = *(const float4*)(p);              // 8 halves: K sub*8 .. sub*8+7
  u.f[1] = *(const float4*)(p + 16);         // 8 halves: K sub*8+16 .. +23
  return u.v;
}

// C[M=1024][ldc] = act(A[M=1024][K=1024] * B[N][K=1024]^T + bias)
// A, B are f32 in memory, converted to f16 while staging to LDS.
// 256 threads = 8 waves; block tile 128x128; per-wave 64x32 (4x2 WMMA tiles).
__global__ __launch_bounds__(256)
void gemm_f16_wmma(const float* __restrict__ A,
                   const float* __restrict__ B,
                   const float* __restrict__ bias,
                   float* __restrict__ C,
                   int N, int ldc, int doTanh)
{
  __shared__ _Float16 As[MT * LDSROW];
  __shared__ _Float16 Bs[NT * LDSROW];

  const int tid  = threadIdx.x;
  const int lane = tid & 31;
  const int wave = tid >> 5;           // 0..7
  const int wm   = wave & 1;           // 2 waves along M  -> 64 rows each
  const int wn   = wave >> 1;          // 4 waves along N  -> 32 cols each
  const int mBlock = blockIdx.x * MT;
  const int nBlock = blockIdx.y * NT;

  v8f acc[4][2] = {};

  for (int k0 = 0; k0 < HID; k0 += KC) {
    // ---- stage A tile: 128 rows x 32 cols, f32 -> f16 (1024 float4 / 256 thr)
    #pragma unroll
    for (int it = 0; it < 4; ++it) {
      int f  = tid + it * 256;         // float4 index in tile
      int r  = f >> 3;                 // row 0..127
      int c4 = f & 7;                  // float4 within the 32-col row
      float4 v = *(const float4*)(A + (size_t)(mBlock + r) * HID + k0 + c4 * 4);
      H4 hh;
      hh.h[0] = (_Float16)v.x; hh.h[1] = (_Float16)v.y;
      hh.h[2] = (_Float16)v.z; hh.h[3] = (_Float16)v.w;
      *(uint2*)&As[r * LDSROW + c4 * 4] = hh.u;
    }
    // ---- stage B tile: rows are W rows (output cols), clamp for the N tail
    #pragma unroll
    for (int it = 0; it < 4; ++it) {
      int f  = tid + it * 256;
      int r  = f >> 3;
      int c4 = f & 7;
      int row = nBlock + r; if (row >= N) row = N - 1;   // clamp, no EXEC change
      float4 v = *(const float4*)(B + (size_t)row * HID + k0 + c4 * 4);
      H4 hh;
      hh.h[0] = (_Float16)v.x; hh.h[1] = (_Float16)v.y;
      hh.h[2] = (_Float16)v.z; hh.h[3] = (_Float16)v.w;
      *(uint2*)&Bs[r * LDSROW + c4 * 4] = hh.u;
    }
    __syncthreads();

    v16h bfrag[2];
    #pragma unroll
    for (int j = 0; j < 2; ++j)
      bfrag[j] = frag_lds(Bs, wn * 32 + j * 16, lane);
    #pragma unroll
    for (int i = 0; i < 4; ++i) {
      v16h afrag = frag_lds(As, wm * 64 + i * 16, lane);
      #pragma unroll
      for (int j = 0; j < 2; ++j)
        acc[i][j] = __builtin_amdgcn_wmma_f32_16x16x32_f16(
            /*neg_a=*/false, afrag, /*neg_b=*/false, bfrag[j],
            /*c_mod=*/(short)0, acc[i][j], /*reuse_a=*/false, /*reuse_b=*/false);
    }
    __syncthreads();
  }

  // ---- epilogue: C/D layout -> lane l holds col (l&15), rows r + 8*(l>>4)
  const int mrow0 = (lane >> 4) * 8;
  const int ncol  = lane & 15;
  #pragma unroll
  for (int i = 0; i < 4; ++i) {
    #pragma unroll
    for (int j = 0; j < 2; ++j) {
      int col = nBlock + wn * 32 + j * 16 + ncol;
      if (col < N) {
        float bv = bias[col];
        #pragma unroll
        for (int r = 0; r < 8; ++r) {
          int mrow = mBlock + wm * 64 + i * 16 + mrow0 + r;
          float val = acc[i][j][r] + bv;
          if (doTanh) val = tanhf(val);
          C[(size_t)mrow * ldc + col] = val;
        }
      }
    }
  }
}

// Fused GRU pointwise: gathers x-gates from W_ih columns (one-hot matmul ==
// column gather), combines with precomputed h_gates (which include b_hh),
// computes h_new = (1-z)*n + z*h, writes it to both h_new and output regions.
__global__ __launch_bounds__(256)
void gru_pointwise(const int* __restrict__ idx,
                   const float* __restrict__ h,       // [1024][1024]
                   const float* __restrict__ W_ih,    // [3072][50000]
                   const float* __restrict__ b_ih,    // [3072]
                   const float* __restrict__ hg,      // [1024][3072] (+b_hh)
                   float* __restrict__ h_new,         // [1024][1024]
                   float* __restrict__ outp)          // [1024][1024]
{
  int t = blockIdx.x * blockDim.x + threadIdx.x;      // 0 .. 1024*1024-1
  int b = t >> 10;
  int j = t & 1023;
  int col = idx[b];
  size_t wb = (size_t)j * CITY + (size_t)col;
  float xr = W_ih[wb]                          + b_ih[j];
  float xz = W_ih[wb + (size_t)HID * CITY]     + b_ih[j + HID];
  float xn = W_ih[wb + (size_t)(2 * HID) * CITY] + b_ih[j + 2 * HID];
  const float* hgb = hg + (size_t)b * G3DIM;
  float hr = hgb[j], hz = hgb[j + HID], hn = hgb[j + 2 * HID];
  float r = 1.0f / (1.0f + __expf(-(xr + hr)));
  float z = 1.0f / (1.0f + __expf(-(xz + hz)));
  float n = tanhf(xn + r * hn);
  float hv = h[t];
  float hnew = (1.0f - z) * n + z * hv;
  h_new[t] = hnew;
  outp[t]  = hnew;
}

extern "C" void kernel_launch(void* const* d_in, const int* in_sizes, int n_in,
                              void* d_out, int out_size, void* d_ws, size_t ws_size,
                              hipStream_t stream) {
  (void)in_sizes; (void)n_in; (void)out_size; (void)ws_size;
  const int*   idx   = (const int*)d_in[0];
  const float* h     = (const float*)d_in[1];   // (1, B, H) flat == [B][H]
  const float* W_ih  = (const float*)d_in[2];
  const float* b_ih  = (const float*)d_in[3];
  const float* W_hh  = (const float*)d_in[4];
  const float* b_hh  = (const float*)d_in[5];
  const float* W_out = (const float*)d_in[6];
  const float* b_out = (const float*)d_in[7];

  float* out   = (float*)d_out;
  float* logit = out;                                       // [1024][50000]
  float* h_new = out + (size_t)BATCH * OUTN;                // [1][1024][1024]
  float* outp  = h_new + (size_t)BATCH * HID;               // [1024][1024]

  float* hg = (float*)d_ws;                                 // [1024][3072] f32

  dim3 blk(256);
  // h_gates = h @ W_hh.T + b_hh   (M=1024, N=3072, K=1024)
  gemm_f16_wmma<<<dim3(BATCH / MT, G3DIM / NT), blk, 0, stream>>>(
      h, W_hh, b_hh, hg, G3DIM, G3DIM, 0);
  // GRU gather + pointwise -> h_new (and duplicate "output")
  gru_pointwise<<<dim3((BATCH * HID) / 256), blk, 0, stream>>>(
      idx, h, W_ih, b_ih, hg, h_new, outp);
  // logit = tanh(h_new @ W_out.T + b_out); grid.x = M-tiles fastest so the
  // 128-col W_out strip is shared by consecutive blocks (single HBM pass).
  gemm_f16_wmma<<<dim3(BATCH / MT, (OUTN + NT - 1) / NT), blk, 0, stream>>>(
      h_new, W_out, b_out, logit, OUTN, OUTN, 1);
}
